// DHMM_55774445306053
// MI455X (gfx1250) — matrix operand order repaired and checked by reading, not compiled, not run on hardware
//
#include <hip/hip_runtime.h>
#include <hip/hip_bf16.h>
#include <math.h>

// ---------------------------------------------------------------------------
// DHMM inference for MI455X (gfx1250): bf16 WMMA GEMM pipeline with
// async-to-LDS double-buffered staging.
// Dims from reference: B=256, T=128, D=256, Z=256, R=512, TR=512, E=512.
// ---------------------------------------------------------------------------

#define BB 256
#define TT 128
#define DD 256
#define ZZ 256
#define RR 512

typedef __attribute__((ext_vector_type(16))) __bf16          v16bf;
typedef __attribute__((ext_vector_type(8)))  float           v8f;
typedef __attribute__((ext_vector_type(8)))  unsigned short  v8u16;
typedef int v4i __attribute__((vector_size(16)));   // matches builtin prototype

#if defined(__has_builtin)
#if __has_builtin(__builtin_amdgcn_global_load_async_to_lds_b128) && \
    __has_builtin(__builtin_amdgcn_s_wait_asynccnt)
#define USE_ASYNC 1
#endif
#endif
#ifndef USE_ASYNC
#define USE_ASYNC 0
#endif

__device__ __forceinline__ unsigned short f2bf(float f) {
    unsigned int u = __float_as_uint(f);
    unsigned int r = u + 0x7FFFu + ((u >> 16) & 1u);   // round-to-nearest-even
    return (unsigned short)(r >> 16);
}
__device__ __forceinline__ float bf2f(unsigned short h) {
    return __uint_as_float(((unsigned int)h) << 16);
}

#if USE_ASYNC
// One 16-byte async DMA copy global -> LDS (ASYNCcnt-tracked on gfx1250).
__device__ __forceinline__ void async_b128(const unsigned short* g, unsigned short* l) {
    __builtin_amdgcn_global_load_async_to_lds_b128(
        (__attribute__((address_space(1))) v4i*)(unsigned long long)(uintptr_t)g,
        (__attribute__((address_space(3))) v4i*)(unsigned int)(uintptr_t)l,
        0, 0);
}
#endif

// ---------------------------------------------------------------------------
// WMMA GEMM:  Y[M,N] = act( A[M,K](bf16, row stride lda) @ W[N,K]^T + bias )
// Block = 4 waves -> 64x64 output tile; K consumed in 32-wide stages staged
// into LDS double buffers via async copies. B-tile is shared by all 4 waves.
// ACT: 0=none  1=relu  2=sigmoid  3=0.5*(tanh(x)+Add)  4=Yf:none / Ybf:relu
// ---------------------------------------------------------------------------
template<int ACT>
__global__ void __launch_bounds__(128)
wmma_gemm(const unsigned short* __restrict__ A, int lda,
          const unsigned short* __restrict__ W,
          const float* __restrict__ bias,
          unsigned short* __restrict__ Ybf,
          float* __restrict__ Yf,
          const unsigned short* __restrict__ Add, int ldadd,
          int M, int N, int K)
{
    __shared__ unsigned short sA[2][64 * 32];   // 64 rows x 32 K, 4KB per stage
    __shared__ unsigned short sB[2][64 * 32];   // 64 cols x 32 K, 4KB per stage

    const int lane  = threadIdx.x;              // 0..31 (wave32)
    const int wave  = threadIdx.y;              // 0..3
    const int tid   = wave * 32 + lane;         // 0..127
    const int mBlk  = blockIdx.y * 64;
    const int mBase = mBlk + wave * 16;
    const int nBase = blockIdx.x * 64;
    const int half  = lane >> 4;
    const int l15   = lane & 15;

    // Staging map: 256 chunks of 8 bf16 (16B) per tile; thread -> chunks tid, tid+128.
    const int r0 = tid >> 2,          q0 = tid & 3;
    const int r1 = (tid + 128) >> 2,  q1 = (tid + 128) & 3;
    const unsigned short* gA0 = A + (size_t)(mBlk  + r0) * (size_t)lda + q0 * 8;
    const unsigned short* gA1 = A + (size_t)(mBlk  + r1) * (size_t)lda + q1 * 8;
    const unsigned short* gB0 = W + (size_t)(nBase + r0) * (size_t)K   + q0 * 8;
    const unsigned short* gB1 = W + (size_t)(nBase + r1) * (size_t)K   + q1 * 8;
    const int dA0 = r0 * 32 + q0 * 8, dA1 = r1 * 32 + q1 * 8;

    auto stage = [&](int buf, int k0) {
#if USE_ASYNC
        async_b128(gA0 + k0, &sA[buf][dA0]);
        async_b128(gA1 + k0, &sA[buf][dA1]);
        async_b128(gB0 + k0, &sB[buf][dA0]);
        async_b128(gB1 + k0, &sB[buf][dA1]);
#else
        *(v8u16*)&sA[buf][dA0] = *(const v8u16*)(gA0 + k0);
        *(v8u16*)&sA[buf][dA1] = *(const v8u16*)(gA1 + k0);
        *(v8u16*)&sB[buf][dA0] = *(const v8u16*)(gB0 + k0);
        *(v8u16*)&sB[buf][dA1] = *(const v8u16*)(gB1 + k0);
#endif
    };

    v8f acc0 = {}, acc1 = {}, acc2 = {}, acc3 = {};

    stage(0, 0);
    int buf = 0;
    for (int k0 = 0; k0 < K; k0 += 32) {
        const bool more = (k0 + 32) < K;
        if (more) stage(buf ^ 1, k0 + 32);     // next stage in flight
#if USE_ASYNC
        if (more) __builtin_amdgcn_s_wait_asynccnt(4);  // keep newest stage's 4 in flight
        else      __builtin_amdgcn_s_wait_asynccnt(0);
#endif
        __syncthreads();                        // current stage visible to all waves

        union { v16bf v; v8u16 h[2]; } af;      // A frag: 16x32, 16 contiguous K per lane (k = 16*half + s)
        const int aOff = (wave * 16 + l15) * 32 + half * 16;
        af.h[0] = *(const v8u16*)&sA[buf][aOff];
        af.h[1] = *(const v8u16*)&sA[buf][aOff + 8];
        // B frags: lane = column, 16 contiguous K values per lane
        v16bf b0 = *(const v16bf*)&sB[buf][(0 * 16 + l15) * 32 + half * 16];
        v16bf b1 = *(const v16bf*)&sB[buf][(1 * 16 + l15) * 32 + half * 16];
        v16bf b2 = *(const v16bf*)&sB[buf][(2 * 16 + l15) * 32 + half * 16];
        v16bf b3 = *(const v16bf*)&sB[buf][(3 * 16 + l15) * 32 + half * 16];

        acc0 = __builtin_amdgcn_wmma_f32_16x16x32_bf16(false, af.v, false, b0, (short)0, acc0, false, false);
        acc1 = __builtin_amdgcn_wmma_f32_16x16x32_bf16(false, af.v, false, b1, (short)0, acc1, false, false);
        acc2 = __builtin_amdgcn_wmma_f32_16x16x32_bf16(false, af.v, false, b2, (short)0, acc2, false, false);
        acc3 = __builtin_amdgcn_wmma_f32_16x16x32_bf16(false, af.v, false, b3, (short)0, acc3, false, false);

        __syncthreads();                        // protect buffer reuse
        buf ^= 1;
    }

    v8f accs[4] = {acc0, acc1, acc2, acc3};
    #pragma unroll
    for (int s = 0; s < 4; ++s) {
        const int col = nBase + s * 16 + l15;
        const float bb = bias[col];
        #pragma unroll
        for (int r = 0; r < 8; ++r) {
            const int row = mBase + half * 8 + r;   // C/D layout: M = r + 8*half
            float xv = accs[s][r] + bb;
            float yv;
            if constexpr (ACT == 1)      yv = fmaxf(xv, 0.0f);
            else if constexpr (ACT == 2) yv = 1.0f / (1.0f + __expf(-xv));
            else if constexpr (ACT == 3) yv = 0.5f * (tanhf(xv) + bf2f(Add[(size_t)row * (size_t)ldadd + col]));
            else                         yv = xv;   // ACT 0 / ACT 4 (pre-act)
            if (Yf)  Yf [(size_t)row * (size_t)N + col] = yv;
            if (Ybf) Ybf[(size_t)row * (size_t)N + col] =
                         f2bf((ACT == 4) ? fmaxf(xv, 0.0f) : yv);
        }
    }
}

// ---------------------------------------------------------------------------
// Elementwise / reduction kernels
// ---------------------------------------------------------------------------
__global__ void f32_to_bf16(const float* __restrict__ src,
                            unsigned short* __restrict__ dst, int n)
{
    int i = blockIdx.x * 256 + threadIdx.x;
    if (i < n) dst[i] = f2bf(src[i]);
}

__global__ void bcast_rows(const float* __restrict__ src, float* dstF,
                           unsigned short* dstB, int cols, int n)
{
    int i = blockIdx.x * 256 + threadIdx.x;
    if (i < n) {
        float v = src[i % cols];
        if (dstF) dstF[i] = v;
        if (dstB) dstB[i] = f2bf(v);
    }
}

// GRU gates: h_new = (1-z)*tanh(in + r*hn) + z*h ; writes fp32 h, bf16 h, rnn[:,t,:]
__global__ void gru_gates(const unsigned short* __restrict__ gi,
                          const unsigned short* __restrict__ gh,
                          float* __restrict__ hF,
                          unsigned short* __restrict__ hBf,
                          unsigned short* __restrict__ rnn, int t)
{
    int i = blockIdx.x * 256 + threadIdx.x;   // over B*R = 131072
    int b = i >> 9;
    int j = i & 511;
    const unsigned short* gib = gi + (size_t)b * 1536;
    const unsigned short* ghb = gh + (size_t)b * 1536;
    float ir = bf2f(gib[j]),       iz = bf2f(gib[512 + j]),  inn = bf2f(gib[1024 + j]);
    float hr = bf2f(ghb[j]),       hz = bf2f(ghb[512 + j]),  hn  = bf2f(ghb[1024 + j]);
    float r  = 1.0f / (1.0f + __expf(-(ir + hr)));
    float zg = 1.0f / (1.0f + __expf(-(iz + hz)));
    float n  = tanhf(inn + r * hn);
    float hnew = (1.0f - zg) * n + zg * hF[i];
    hF[i] = hnew;
    unsigned short hb = f2bf(hnew);
    hBf[i] = hb;
    rnn[((size_t)b * TT + t) * RR + j] = hb;
}

// In-place per-sequence reversal of rnn (reversal is an involution).
__global__ void reverse_rows(unsigned short* __restrict__ rnn,
                             const int* __restrict__ lens)
{
    int b = blockIdx.x, t = blockIdx.y, j = threadIdx.x;   // grid (B, T/2), block R
    int len = lens[b];
    int tt  = len - 1 - t;
    if (t < tt) {
        size_t i0 = ((size_t)b * TT + t)  * RR + j;
        size_t i1 = ((size_t)b * TT + tt) * RR + j;
        unsigned short a = rnn[i0]; rnn[i0] = rnn[i1]; rnn[i1] = a;
    }
}

__global__ void prior_mix(const float* __restrict__ gate,
                          const float* __restrict__ mulin,
                          const float* __restrict__ prop,
                          float* __restrict__ pmu)
{
    int i = blockIdx.x * 256 + threadIdx.x;   // B*Z = 65536
    float g = gate[i];
    pmu[i] = (1.0f - g) * mulin[i] + g * prop[i];
}

// z_t = mu + eps*exp(lv/2); kl row-sum per batch element (block reduce, no atomics)
__global__ void z_and_kl(const float* __restrict__ mu, const float* __restrict__ lv,
                         const float* __restrict__ pmu, const float* __restrict__ plv,
                         const float* __restrict__ eps_t,
                         unsigned short* __restrict__ zbf,
                         float* __restrict__ klmat, int t)
{
    int b = blockIdx.x, z = threadIdx.x;      // grid B, block Z=256
    int i = b * ZZ + z;
    float m = mu[i], l = lv[i], pm = pmu[i], pl = plv[i];
    float zt = m + eps_t[(size_t)b * ZZ + z] * __expf(0.5f * l);
    zbf[i] = f2bf(zt);
    float dm = m - pm;
    float kle = 0.5f * (pl - l + (__expf(l) + dm * dm) * __expf(-pl) - 1.0f);
    __shared__ float red[256];
    red[z] = kle; __syncthreads();
    for (int s = 128; s > 0; s >>= 1) { if (z < s) red[z] += red[z + s]; __syncthreads(); }
    if (z == 0) klmat[b * TT + t] = red[0];
}

// BCE-with-logits per-row sum (block reduce per batch element)
__global__ void bce_row(const float* __restrict__ logits,
                        const float* __restrict__ x, int t,
                        float* __restrict__ browAll)
{
    int b = blockIdx.x, d = threadIdx.x;      // grid B, block D=256
    float l  = logits[b * DD + d];
    float xv = x[((size_t)b * TT + t) * DD + d];
    float v  = fmaxf(l, 0.0f) - l * xv + log1pf(__expf(-fabsf(l)));
    __shared__ float red[256];
    red[d] = v; __syncthreads();
    for (int s = 128; s > 0; s >>= 1) { if (d < s) red[d] += red[d + s]; __syncthreads(); }
    if (d == 0) browAll[(size_t)t * BB + b] = red[0];
}

// Final masked means -> out[0]=rec_loss, out[1]=kl_loss
__global__ void finalize_kernel(const float* __restrict__ klmat,
                                const float* __restrict__ browAll,
                                const int* __restrict__ lens,
                                float* __restrict__ out)
{
    __shared__ float red[256];
    __shared__ int   slen[256];
    int tid = threadIdx.x;
    slen[tid] = lens[tid];
    __syncthreads();

    // kl: thread = batch element b, sum over t < len_b
    float klp = 0.0f;
    int lb = slen[tid];
    for (int t = 0; t < lb; ++t) klp += klmat[tid * TT + t];
    red[tid] = klp; __syncthreads();
    for (int s = 128; s > 0; s >>= 1) { if (tid < s) red[tid] += red[tid + s]; __syncthreads(); }
    float klsum = red[0]; __syncthreads();

    // denom = sum(len_b)
    red[tid] = (float)slen[tid]; __syncthreads();
    for (int s = 128; s > 0; s >>= 1) { if (tid < s) red[tid] += red[tid + s]; __syncthreads(); }
    float denom = red[0]; __syncthreads();

    // rec: thread = timestep t (<128): bce_t * count(len_b > t)
    float rp = 0.0f;
    if (tid < TT) {
        float bsum = 0.0f;
        for (int b = 0; b < BB; ++b) bsum += browAll[(size_t)tid * BB + b];
        float bce = bsum * (1.0f / (float)(BB * DD));
        int cnt = 0;
        for (int b = 0; b < BB; ++b) cnt += (tid < slen[b]) ? 1 : 0;
        rp = bce * (float)cnt;
    }
    red[tid] = rp; __syncthreads();
    for (int s = 128; s > 0; s >>= 1) { if (tid < s) red[tid] += red[tid + s]; __syncthreads(); }
    if (tid == 0) { out[0] = red[0] / denom; out[1] = klsum / denom; }
}

// ---------------------------------------------------------------------------
// Host-side GEMM dispatcher
// ---------------------------------------------------------------------------
static void gemm(int act,
                 const unsigned short* A, int lda,
                 const unsigned short* W, const float* bias,
                 unsigned short* Ybf, float* Yf,
                 const unsigned short* Add, int ldadd,
                 int M, int N, int K, hipStream_t s)
{
    dim3 grid(N / 64, M / 64), block(32, 4);
    switch (act) {
    case 0: wmma_gemm<0><<<grid, block, 0, s>>>(A, lda, W, bias, Ybf, Yf, Add, ldadd, M, N, K); break;
    case 1: wmma_gemm<1><<<grid, block, 0, s>>>(A, lda, W, bias, Ybf, Yf, Add, ldadd, M, N, K); break;
    case 2: wmma_gemm<2><<<grid, block, 0, s>>>(A, lda, W, bias, Ybf, Yf, Add, ldadd, M, N, K); break;
    case 3: wmma_gemm<3><<<grid, block, 0, s>>>(A, lda, W, bias, Ybf, Yf, Add, ldadd, M, N, K); break;
    default: wmma_gemm<4><<<grid, block, 0, s>>>(A, lda, W, bias, Ybf, Yf, Add, ldadd, M, N, K); break;
    }
}

extern "C" void kernel_launch(void* const* d_in, const int* in_sizes, int n_in,
                              void* d_out, int out_size, void* d_ws, size_t ws_size,
                              hipStream_t stream)
{
    (void)in_sizes; (void)n_in; (void)out_size; (void)ws_size;

    const float* x      = (const float*)d_in[0];   // [B,T,D]
    const float* x_rev  = (const float*)d_in[1];   // [B,T,D]
    const int*   x_lens = (const int*)  d_in[2];   // [B]
    const float* eps    = (const float*)d_in[3];   // [T,B,Z]
    const float* Wih  = (const float*)d_in[4];  const float* bih  = (const float*)d_in[5];
    const float* Whh  = (const float*)d_in[6];  const float* bhh  = (const float*)d_in[7];
    const float* h_0  = (const float*)d_in[8];
    const float* zq0  = (const float*)d_in[9];
    const float* Wg1  = (const float*)d_in[10]; const float* bg1  = (const float*)d_in[11];
    const float* Wg2  = (const float*)d_in[12]; const float* bg2  = (const float*)d_in[13];
    const float* Wp1  = (const float*)d_in[14]; const float* bp1  = (const float*)d_in[15];
    const float* Wp2  = (const float*)d_in[16]; const float* bp2  = (const float*)d_in[17];
    const float* Wmu  = (const float*)d_in[18]; const float* bmu  = (const float*)d_in[19];
    const float* Wlv  = (const float*)d_in[20]; const float* blv  = (const float*)d_in[21];
    const float* Wzh  = (const float*)d_in[22]; const float* bzh  = (const float*)d_in[23];
    const float* Whmu = (const float*)d_in[24]; const float* bhmu = (const float*)d_in[25];
    const float* Whlv = (const float*)d_in[26]; const float* bhlv = (const float*)d_in[27];
    const float* We1  = (const float*)d_in[28]; const float* be1  = (const float*)d_in[29];
    const float* We2  = (const float*)d_in[30]; const float* be2  = (const float*)d_in[31];
    const float* We3  = (const float*)d_in[32]; const float* be3  = (const float*)d_in[33];

    // ---- workspace carve-up ----
    char* base = (char*)d_ws;
    size_t off = 0;
    auto alloc = [&](size_t bytes) -> void* {
        off = (off + 255) & ~(size_t)255;
        void* p = base + off; off += bytes; return p;
    };
    auto aU = [&](size_t n) { return (unsigned short*)alloc(n * 2); };
    auto aF = [&](size_t n) { return (float*)alloc(n * 4); };

    unsigned short *wIhB = aU(1536*256), *wHhB = aU(1536*512);
    unsigned short *wG1B = aU(512*256),  *wG2B = aU(256*512);
    unsigned short *wP1B = aU(512*256),  *wP2B = aU(256*512);
    unsigned short *wMuB = aU(256*256),  *wLvB = aU(256*256);
    unsigned short *wZhB = aU(512*256),  *wHmuB = aU(256*512), *wHlvB = aU(256*512);
    unsigned short *wE1B = aU(512*256),  *wE2B = aU(512*512),  *wE3B = aU(256*512);
    unsigned short *xrevB = aU((size_t)BB*TT*DD);
    unsigned short *rnn   = aU((size_t)BB*TT*RR);      // GRU outputs (reversed in place)
    unsigned short *hBf   = aU(BB*RR);
    unsigned short *gi    = aU(BB*1536), *gh = aU(BB*1536);
    unsigned short *zBf   = aU(BB*ZZ);
    unsigned short *g1    = aU(BB*512), *p1 = aU(BB*512);
    unsigned short *propR = aU(BB*ZZ);
    unsigned short *hcomb = aU(BB*512);
    unsigned short *h1b   = aU(BB*512), *h2b = aU(BB*512);
    float *hF    = aF(BB*RR);
    float *gateF = aF(BB*ZZ), *propF = aF(BB*ZZ), *mulin = aF(BB*ZZ);
    float *pmuF  = aF(BB*ZZ), *plvF  = aF(BB*ZZ);
    float *muF   = aF(BB*ZZ), *lvF   = aF(BB*ZZ);
    float *logit = aF(BB*DD);
    float *klmat = aF((size_t)BB*TT);
    float *browA = aF((size_t)TT*BB);

    // ---- 1. convert weights + x_rev to bf16 ----
    struct CV { const float* s; unsigned short* d; int n; };
    const CV cvs[] = {
        {Wih, wIhB, 1536*256}, {Whh, wHhB, 1536*512},
        {Wg1, wG1B, 512*256},  {Wg2, wG2B, 256*512},
        {Wp1, wP1B, 512*256},  {Wp2, wP2B, 256*512},
        {Wmu, wMuB, 256*256},  {Wlv, wLvB, 256*256},
        {Wzh, wZhB, 512*256},  {Whmu, wHmuB, 256*512}, {Whlv, wHlvB, 256*512},
        {We1, wE1B, 512*256},  {We2, wE2B, 512*512},   {We3, wE3B, 256*512},
        {x_rev, xrevB, BB*TT*DD},
    };
    for (const CV& c : cvs)
        f32_to_bf16<<<(c.n + 255) / 256, 256, 0, stream>>>(c.s, c.d, c.n);

    // ---- 2. init h0 / z0 broadcasts ----
    bcast_rows<<<(BB*RR + 255)/256, 256, 0, stream>>>(h_0, hF, hBf, RR, BB*RR);
    bcast_rows<<<(BB*ZZ + 255)/256, 256, 0, stream>>>(zq0, nullptr, zBf, ZZ, BB*ZZ);

    // ---- 3. GRU over x_rev (sequential) ----
    for (int t = 0; t < TT; ++t) {
        // gi = x_rev[:,t,:] @ Wih^T + bih      (A row stride T*D)
        gemm(0, xrevB + (size_t)t * DD, TT*DD, wIhB, bih, gi, nullptr, nullptr, 0,
             BB, 1536, DD, stream);
        // gh = h @ Whh^T + bhh
        gemm(0, hBf, RR, wHhB, bhh, gh, nullptr, nullptr, 0, BB, 1536, RR, stream);
        gru_gates<<<(BB*RR)/256, 256, 0, stream>>>(gi, gh, hF, hBf, rnn, t);
    }

    // ---- 4. reverse rnn per sequence (in place; reversal is involutive) ----
    reverse_rows<<<dim3(BB, TT/2), RR, 0, stream>>>(rnn, x_lens);

    // ---- 5. scan over T ----
    for (int t = 0; t < TT; ++t) {
        // GatedTransition
        gemm(1, zBf, ZZ, wG1B, bg1, g1, nullptr, nullptr, 0, BB, 512, ZZ, stream);       // relu
        gemm(2, g1, 512, wG2B, bg2, nullptr, gateF, nullptr, 0, BB, ZZ, 512, stream);    // sigmoid
        gemm(1, zBf, ZZ, wP1B, bp1, p1, nullptr, nullptr, 0, BB, 512, ZZ, stream);       // relu
        gemm(4, p1, 512, wP2B, bp2, propR, propF, nullptr, 0, BB, ZZ, 512, stream);      // raw + relu-bf16
        gemm(0, zBf, ZZ, wMuB, bmu, nullptr, mulin, nullptr, 0, BB, ZZ, ZZ, stream);
        prior_mix<<<(BB*ZZ)/256, 256, 0, stream>>>(gateF, mulin, propF, pmuF);
        gemm(0, propR, ZZ, wLvB, blv, nullptr, plvF, nullptr, 0, BB, ZZ, ZZ, stream);
        // PostNet: h_comb = 0.5*(tanh(z@Wzh^T+bzh) + rnn[:,t,:])
        gemm(3, zBf, ZZ, wZhB, bzh, hcomb, nullptr, rnn + (size_t)t * RR, TT*RR,
             BB, RR, ZZ, stream);
        gemm(0, hcomb, RR, wHmuB, bhmu, nullptr, muF, nullptr, 0, BB, ZZ, RR, stream);
        gemm(0, hcomb, RR, wHlvB, bhlv, nullptr, lvF, nullptr, 0, BB, ZZ, RR, stream);
        z_and_kl<<<BB, ZZ, 0, stream>>>(muF, lvF, pmuF, plvF,
                                        eps + (size_t)t * BB * ZZ, zBf, klmat, t);
        // Emitter
        gemm(1, zBf, ZZ, wE1B, be1, h1b, nullptr, nullptr, 0, BB, 512, ZZ, stream);
        gemm(1, h1b, 512, wE2B, be2, h2b, nullptr, nullptr, 0, BB, 512, 512, stream);
        gemm(0, h2b, 512, wE3B, be3, nullptr, logit, nullptr, 0, BB, DD, 512, stream);
        bce_row<<<BB, DD, 0, stream>>>(logit, x, t, browA);
    }

    // ---- 6. masked means ----
    finalize_kernel<<<1, 256, 0, stream>>>(klmat, browA, x_lens, (float*)d_out);
}
